// Decoder2_47442208751775
// MI455X (gfx1250) — matrix operand order repaired and checked by reading, not compiled, not run on hardware
//
#include <hip/hip_runtime.h>
#include <stdint.h>

// Problem constants (from reference): B=8192, T=96, F=64, H=256, 3H=768
#define B_   8192
#define T_   96
#define F_   64
#define H_   256
#define H3_  768
#define TB   32      // batch rows per workgroup (persistent tile)

typedef __bf16 bf16x16 __attribute__((ext_vector_type(16)));
typedef float  f32x8   __attribute__((ext_vector_type(8)));

struct __align__(16) U128 { unsigned int x, y, z, w; };
struct __align__(16) F128 { float x, y, z, w; };

union ABfrag { U128 q[2]; bf16x16 v; };   // 32 bytes: one WMMA A/B operand per lane
union PK8   { U128 q; unsigned short s[8]; };

static __device__ __forceinline__ unsigned short f2bf(float f) {
    unsigned u = __builtin_bit_cast(unsigned, f);
    u += 0x7FFFu + ((u >> 16) & 1u);            // round-to-nearest-even
    return (unsigned short)(u >> 16);
}
static __device__ __forceinline__ float sigmoidf_(float x) {
    return 1.0f / (1.0f + __expf(-x));
}
static __device__ __forceinline__ float tanhf_(float x) {
    float e = __expf(-2.0f * x);
    return (1.0f - e) / (1.0f + e);
}

// ---------------------------------------------------------------------------
// Prologue: pack W_in (kernel rows 1..64) and W_r into WMMA B-operand layout.
// B 32x16 bf16 fragment: lane L, element e -> K = (L/16)*8 + e for e<8, else 16 + (L/16)*8 + (e-8), N = L%16.
// Block (kt, ntg) stored contiguously: 32 lanes * 32B = 1KB per block.
// ---------------------------------------------------------------------------
__global__ void pack_weights_kernel(const float* __restrict__ kern,
                                    const float* __restrict__ rker,
                                    unsigned short* __restrict__ winP,
                                    unsigned short* __restrict__ wrP) {
    const int bi = blockIdx.x;          // 0..95 -> W_in blocks, 96..479 -> W_r
    const int L  = threadIdx.x;         // 0..31
    const int half = L >> 4, l16 = L & 15;
    if (bi < 96) {
        const int kt = bi / 48, ntg = bi % 48;
        unsigned short* dst = winP + bi * 512 + L * 16;
        const int N = ntg * 16 + l16;
#pragma unroll
        for (int e = 0; e < 16; ++e) {
            const int K = kt * 32 + (e < 8 ? half * 8 + e : 16 + half * 8 + (e - 8)); // feature index 0..63
            dst[e] = f2bf(kern[(size_t)(1 + K) * H3_ + N]); // rows 1..64 of kernel
        }
    } else {
        const int b2 = bi - 96;
        const int kt = b2 / 48, ntg = b2 % 48;
        unsigned short* dst = wrP + b2 * 512 + L * 16;
        const int N = ntg * 16 + l16;
#pragma unroll
        for (int e = 0; e < 16; ++e) {
            const int K = kt * 32 + (e < 8 ? half * 8 + e : 16 + half * 8 + (e - 8)); // hidden index 0..255
            dst[e] = f2bf(rker[(size_t)K * H3_ + N]);
        }
    }
}

// ---------------------------------------------------------------------------
// Persistent GRU kernel. One WG = 32 batch rows, 8 waves * 32 lanes.
// Wave w owns output columns [32w, 32w+32) of each gate (z, r, cand).
// h kept in f32 registers (WMMA C/D layout) across all 96 steps.
// Weight fragments are re-loaded from L2 every step; a laundered SGPR
// *offset* (not pointer) keeps the loads un-hoistable while preserving
// global address-space provenance -> global_load_b128, LOADcnt only.
// ---------------------------------------------------------------------------
__global__ __launch_bounds__(256) void gru_persistent_kernel(
    const float* __restrict__ feat,        // [B,T,F]
    const float* __restrict__ init_state,  // [B,H]
    const float* __restrict__ init_inp,    // [B,1]
    const float* __restrict__ kern,        // [1+F,3H] (row 0 used: prev_out rank-1)
    const float* __restrict__ ibias,       // [3H]
    const float* __restrict__ rbias,       // [3H]
    const float* __restrict__ dwv,         // [H]
    const float* __restrict__ dbp,         // [1]
    const U128*  __restrict__ winP,        // packed W_in  (2x48 blocks)
    const U128*  __restrict__ wrP,         // packed W_r   (8x48 blocks)
    float* __restrict__ out)               // [B,T]
{
    __shared__ __align__(16) unsigned short hS[TB * H_];   // h tile, bf16 row-major
    __shared__ __align__(16) unsigned short fS[TB * F_];   // feature tile, bf16
    __shared__ float poS[TB];                              // prev scalar output
    __shared__ float outP[8][TB];                          // per-wave output partials

    const int tid  = threadIdx.x;
    const int w    = tid >> 5;
    const int L    = tid & 31;
    const int half = L >> 4, l16 = L & 15;
    const int b0   = blockIdx.x * TB;

    // Per-lane column constants for this wave's gate slabs.
    float k0c[3][2], dw[2];
    float bz[2], br[2], ibc2[2], rbc2[2];   // combined biases for z/r; split for cand
#pragma unroll
    for (int nt = 0; nt < 2; ++nt) {
        const int c0 = w * 32 + nt * 16 + l16;
        k0c[0][nt] = kern[0 * H_ + c0];
        k0c[1][nt] = kern[1 * H_ + c0];
        k0c[2][nt] = kern[2 * H_ + c0];
        bz[nt]   = ibias[0 * H_ + c0] + rbias[0 * H_ + c0];
        br[nt]   = ibias[1 * H_ + c0] + rbias[1 * H_ + c0];
        ibc2[nt] = ibias[2 * H_ + c0];
        rbc2[nt] = rbias[2 * H_ + c0];
        dw[nt]   = dwv[c0];
    }
    const float db = dbp[0];

    // Load initial hidden state into f32 registers (D-tile layout) + bf16 LDS.
    f32x8 h[2][2];
#pragma unroll
    for (int nt = 0; nt < 2; ++nt)
#pragma unroll
        for (int mt = 0; mt < 2; ++mt)
#pragma unroll
            for (int v = 0; v < 8; ++v) {
                const int m = mt * 16 + v + 8 * half;
                const int c = w * 32 + nt * 16 + l16;
                const float hv = init_state[(size_t)(b0 + m) * H_ + c];
                h[nt][mt][v] = hv;
                hS[m * H_ + c] = f2bf(hv);
            }
    if (tid < TB) poS[tid] = init_inp[b0 + tid];

    for (int t = 0; t < T_; ++t) {
        // Launder a uniform 32-bit offset (value is always 0, but the compiler
        // cannot prove it): weight loads stay inside the t-loop and keep
        // global address-space provenance (global_load_b128, not flat).
        unsigned int woff = 0;
        asm volatile("" : "+s"(woff));
        const U128* winp_t = winP + woff;
        const U128* wrp_t  = wrP  + woff;

        // ---- stage feature tile (32x64 f32 -> bf16 LDS), fully coalesced ----
        {
            const int row = tid >> 3, grp = tid & 7;
            const float* fp = feat + ((size_t)(b0 + row) * T_ + t) * F_ + grp * 8;
            const F128 f0 = ((const F128*)fp)[0];
            const F128 f1 = ((const F128*)fp)[1];
            PK8 o;
            o.s[0]=f2bf(f0.x); o.s[1]=f2bf(f0.y); o.s[2]=f2bf(f0.z); o.s[3]=f2bf(f0.w);
            o.s[4]=f2bf(f1.x); o.s[5]=f2bf(f1.y); o.s[6]=f2bf(f1.z); o.s[7]=f2bf(f1.w);
            *((U128*)(fS + row * F_ + grp * 8)) = o.q;
        }
        __syncthreads();   // bar1: fS, hS (t-1), poS (t-1) visible

        // ---- accumulators: 0=z, 1=r, 2=cand(mx part), 3=cand(mh part) ----
        f32x8 acc[4][2][2];
        const f32x8 z8 = {0.f,0.f,0.f,0.f,0.f,0.f,0.f,0.f};
#pragma unroll
        for (int g = 0; g < 4; ++g)
#pragma unroll
            for (int nt = 0; nt < 2; ++nt)
#pragma unroll
                for (int mt = 0; mt < 2; ++mt) acc[g][nt][mt] = z8;

        // ---- mx = feat @ W_in  (K = 64 -> kt 0..1) ----
#pragma unroll
        for (int kt = 0; kt < 2; ++kt) {
            ABfrag a[2];
#pragma unroll
            for (int mt = 0; mt < 2; ++mt) {
                // 16-bit A layout: elems 0..7 at K=half*8+e, 8..15 at K=16+half*8+e
                const U128* p = (const U128*)(fS + (mt * 16 + l16) * F_ + kt * 32 + half * 8);
                a[mt].q[0] = p[0]; a[mt].q[1] = p[2];
            }
#pragma unroll
            for (int g = 0; g < 3; ++g)
#pragma unroll
                for (int nt = 0; nt < 2; ++nt) {
                    const int ntg = g * 16 + w * 2 + nt;
                    ABfrag b;
                    const U128* bp = winp_t + (size_t)(kt * 48 + ntg) * 64 + L * 2;
                    b.q[0] = bp[0]; b.q[1] = bp[1];
#pragma unroll
                    for (int mt = 0; mt < 2; ++mt)
                        acc[g][nt][mt] = __builtin_amdgcn_wmma_f32_16x16x32_bf16(
                            false, a[mt].v, false, b.v, (short)0, acc[g][nt][mt], false, false);
                }
        }

        // ---- mh = h @ W_r  (K = 256 -> kt 0..7); cand part kept separate ----
#pragma unroll
        for (int kt = 0; kt < 8; ++kt) {
            ABfrag a[2];
#pragma unroll
            for (int mt = 0; mt < 2; ++mt) {
                const U128* p = (const U128*)(hS + (mt * 16 + l16) * H_ + kt * 32 + half * 8);
                a[mt].q[0] = p[0]; a[mt].q[1] = p[2];
            }
#pragma unroll
            for (int g = 0; g < 3; ++g)
#pragma unroll
                for (int nt = 0; nt < 2; ++nt) {
                    const int ntg = g * 16 + w * 2 + nt;
                    ABfrag b;
                    const U128* bp = wrp_t + (size_t)(kt * 48 + ntg) * 64 + L * 2;
                    b.q[0] = bp[0]; b.q[1] = bp[1];
                    const int ga = (g == 2) ? 3 : g;
#pragma unroll
                    for (int mt = 0; mt < 2; ++mt)
                        acc[ga][nt][mt] = __builtin_amdgcn_wmma_f32_16x16x32_bf16(
                            false, a[mt].v, false, b.v, (short)0, acc[ga][nt][mt], false, false);
                }
        }

        // ---- rank-1 prev_out term, gates, f32 state update, output partials ----
        float pout[2][8];
#pragma unroll
        for (int mt = 0; mt < 2; ++mt)
#pragma unroll
            for (int v = 0; v < 8; ++v) pout[mt][v] = 0.f;

#pragma unroll
        for (int mt = 0; mt < 2; ++mt)
#pragma unroll
            for (int v = 0; v < 8; ++v) {
                const float p = poS[mt * 16 + v + 8 * half];
#pragma unroll
                for (int nt = 0; nt < 2; ++nt) {
                    const float z  = sigmoidf_(acc[0][nt][mt][v] + p * k0c[0][nt] + bz[nt]);
                    const float r  = sigmoidf_(acc[1][nt][mt][v] + p * k0c[1][nt] + br[nt]);
                    const float hh = acc[3][nt][mt][v] + rbc2[nt];   // recurrent bias inside r*
                    const float cx = acc[2][nt][mt][v] + p * k0c[2][nt] + ibc2[nt];
                    const float cd = tanhf_(cx + r * hh);
                    float hv = h[nt][mt][v];
                    hv = z * hv + (1.0f - z) * cd;
                    h[nt][mt][v] = hv;
                    pout[mt][v] += hv * dw[nt];
                }
            }

        __syncthreads();   // bar2: all hS/fS reads of this step are done

        // write updated hidden state back to LDS as bf16 (next step's A operand)
#pragma unroll
        for (int nt = 0; nt < 2; ++nt)
#pragma unroll
            for (int mt = 0; mt < 2; ++mt)
#pragma unroll
                for (int v = 0; v < 8; ++v)
                    hS[(mt * 16 + v + 8 * half) * H_ + w * 32 + nt * 16 + l16] =
                        f2bf(h[nt][mt][v]);

        // reduce this wave's 32-column contribution to the output dot (wave32)
#pragma unroll
        for (int mt = 0; mt < 2; ++mt)
#pragma unroll
            for (int v = 0; v < 8; ++v) {
                float pv = pout[mt][v];
                pv += __shfl_xor(pv, 1, 32);
                pv += __shfl_xor(pv, 2, 32);
                pv += __shfl_xor(pv, 4, 32);
                pv += __shfl_xor(pv, 8, 32);
                if (l16 == 0) outP[w][mt * 16 + v + 8 * half] = pv;
            }

        __syncthreads();   // bar3: outP, hS complete

        if (tid < TB) {
            float o = db;
#pragma unroll
            for (int ww = 0; ww < 8; ++ww) o += outP[ww][tid];
            poS[tid] = o;                                   // prev_out for t+1
            out[(size_t)(b0 + tid) * T_ + t] = o;
        }
    }
}

extern "C" void kernel_launch(void* const* d_in, const int* in_sizes, int n_in,
                              void* d_out, int out_size, void* d_ws, size_t ws_size,
                              hipStream_t stream) {
    (void)in_sizes; (void)n_in; (void)out_size; (void)ws_size;
    const float* feat  = (const float*)d_in[0];
    const float* hinit = (const float*)d_in[1];
    const float* oinit = (const float*)d_in[2];
    const float* kern  = (const float*)d_in[3];
    const float* rker  = (const float*)d_in[4];
    const float* ibias = (const float*)d_in[5];
    const float* rbias = (const float*)d_in[6];
    const float* dw    = (const float*)d_in[7];
    const float* dbias = (const float*)d_in[8];

    unsigned short* winP = (unsigned short*)d_ws;        // 96 blocks * 1KB
    unsigned short* wrP  = winP + 96 * 512;              // 384 blocks * 1KB

    pack_weights_kernel<<<480, 32, 0, stream>>>(kern, rker, winP, wrP);
    gru_persistent_kernel<<<B_ / TB, 256, 0, stream>>>(
        feat, hinit, oinit, kern, ibias, rbias, dw, dbias,
        (const U128*)winP, (const U128*)wrP, (float*)d_out);
}
